// ConditionalLoRALinear_25752623907395
// MI455X (gfx1250) — matrix operand order repaired and checked by reading, not compiled, not run on hardware
//
#include <hip/hip_runtime.h>

// ---------------------------------------------------------------------------
// ConditionalLoRALinear for MI455X (gfx1250, wave32, WMMA)
//   out = x @ W^T + b + mask * 2.0 * ((x @ A^T) @ B^T)
// M=16384 (B*S), N=4096, K=4096, R=8
// Primary path: operands pre-split to bf16 hi/lo in ws; GEMM hot loop uses
// GLOBAL_LOAD_ASYNC_TO_LDS_B128 (ASYNCcnt) into double-buffered LDS, then
// 3x v_wmma_f32_16x16x32_bf16 per tile with fp32 accumulate. No VGPR staging.
// ---------------------------------------------------------------------------

#define M_TOT 16384
#define N_TOT 4096
#define K_TOT 4096
#define R_LORA 8
#define SCALING 2.0f
#define COMP_TOKEN 7

#define BM 128
#define BN 128
#define BK 32
#define LDK 40                   // padded bf16 K-stride (80 B), conflict-free b128
#define KT (K_TOT / BK)          // 128
#define TILE_E (BM * LDK)        // 5120 u16 per tile
#define TILE_B (TILE_E * 2)      // 10240 bytes per tile

typedef unsigned short u16;
typedef __attribute__((ext_vector_type(16))) __bf16 v16bf;
typedef __attribute__((ext_vector_type(8)))  __bf16 v8bf;
typedef __attribute__((ext_vector_type(8)))  float  v8f;

__device__ __forceinline__ unsigned int pack_bf16_rne(float a, float b) {
#if __has_builtin(__builtin_amdgcn_cvt_pk_bf16_f32)
    typedef __attribute__((ext_vector_type(2))) __bf16 v2bf;
    v2bf v = __builtin_amdgcn_cvt_pk_bf16_f32(a, b);
    return __builtin_bit_cast(unsigned int, v);
#else
    unsigned int ua = __float_as_uint(a), ub = __float_as_uint(b);
    unsigned int ra = (ua + 0x7FFFu + ((ua >> 16) & 1u)) >> 16;
    unsigned int rb = (ub + 0x7FFFu + ((ub >> 16) & 1u)) >> 16;
    return (ra & 0xFFFFu) | (rb << 16);
#endif
}

__device__ __forceinline__ void split4(float4 v, uint2& hv, uint2& lv) {
    hv.x = pack_bf16_rne(v.x, v.y);
    hv.y = pack_bf16_rne(v.z, v.w);
    float r0 = v.x - __uint_as_float(hv.x << 16);
    float r1 = v.y - __uint_as_float(hv.x & 0xFFFF0000u);
    float r2 = v.z - __uint_as_float(hv.y << 16);
    float r3 = v.w - __uint_as_float(hv.y & 0xFFFF0000u);
    lv.x = pack_bf16_rne(r0, r1);
    lv.y = pack_bf16_rne(r2, r3);
}

// Async copy 16B global -> LDS (per lane). Tracked by ASYNCcnt.
__device__ __forceinline__ void async_copy_b128(unsigned lds_byte_off, const u16* g) {
    asm volatile("global_load_async_to_lds_b128 %0, %1, off"
                 :: "v"(lds_byte_off), "v"(g) : "memory");
}
__device__ __forceinline__ void wait_async0() {
    asm volatile("s_wait_asynccnt 0x0" ::: "memory");
}

// Fragment builders (LDS layouts match ISA §7.12.2 for 16-bit A and B).
__device__ __forceinline__ v16bf loadA_frag(const u16* s, int wm, int mt, int lg, int lh) {
    const u16* p = s + (wm + mt * 16 + lg) * LDK + lh * 8;
    v8bf a0 = *(const v8bf*)p;
    v8bf a1 = *(const v8bf*)(p + 16);
    return __builtin_shufflevector(a0, a1, 0, 1, 2, 3, 4, 5, 6, 7,
                                           8, 9, 10, 11, 12, 13, 14, 15);
}
__device__ __forceinline__ v16bf loadB_frag(const u16* s, int wn, int nt, int lg, int lh) {
    const u16* p = s + (wn + nt * 16 + lg) * LDK + lh * 8;
    v8bf b0 = *(const v8bf*)p;
    v8bf b1 = *(const v8bf*)(p + 16);
    return __builtin_shufflevector(b0, b1, 0, 1, 2, 3, 4, 5, 6, 7,
                                           8, 9, 10, 11, 12, 13, 14, 15);
}

// ---------------------------------------------------------------------------
// One-time elementwise split: fp32 -> (bf16 hi, bf16 lo). 8 floats/thread.
// ---------------------------------------------------------------------------
__global__ __launch_bounds__(256) void split_bf16_kernel(
    const float* __restrict__ in, u16* __restrict__ hi, u16* __restrict__ lo,
    long long n8) {
    long long i = (long long)blockIdx.x * 256 + threadIdx.x;
    if (i >= n8) return;
    size_t off = (size_t)i * 8;
    float4 a = *(const float4*)(in + off);
    float4 b = *(const float4*)(in + off + 4);
    uint2 ha, la, hb, lb;
    split4(a, ha, la);
    split4(b, hb, lb);
    uint4 h; h.x = ha.x; h.y = ha.y; h.z = hb.x; h.w = hb.y;
    uint4 l; l.x = la.x; l.y = la.y; l.z = lb.x; l.w = lb.y;
    *(uint4*)(hi + off) = h;
    *(uint4*)(lo + off) = l;
}

// ---------------------------------------------------------------------------
// Pass 1: t[m][r] = gate(m) * SCALING * sum_k x[m,k] * loraA[r,k]
// ---------------------------------------------------------------------------
__global__ __launch_bounds__(256) void lora_t_kernel(
    const float* __restrict__ x, const int* __restrict__ ids,
    const float* __restrict__ lA, float* __restrict__ t) {
    const int m = blockIdx.x;
    const int tid = threadIdx.x;
    const float* xr = x + (size_t)m * K_TOT;

    float acc[R_LORA];
#pragma unroll
    for (int r = 0; r < R_LORA; ++r) acc[r] = 0.0f;

    for (int k0 = tid * 4; k0 < K_TOT; k0 += 256 * 4) {
        float4 xv = *(const float4*)(xr + k0);
#pragma unroll
        for (int r = 0; r < R_LORA; ++r) {
            float4 av = *(const float4*)(lA + r * K_TOT + k0);
            acc[r] += xv.x * av.x + xv.y * av.y + xv.z * av.z + xv.w * av.w;
        }
    }

    __shared__ float red[R_LORA][256];
#pragma unroll
    for (int r = 0; r < R_LORA; ++r) red[r][tid] = acc[r];
    __syncthreads();
    for (int s = 128; s > 0; s >>= 1) {
        if (tid < s) {
#pragma unroll
            for (int r = 0; r < R_LORA; ++r) red[r][tid] += red[r][tid + s];
        }
        __syncthreads();
    }
    if (tid < R_LORA) {
        float gate = (ids[m] == COMP_TOKEN) ? SCALING : 0.0f;
        t[(size_t)m * R_LORA + tid] = gate * red[tid][0];
    }
}

// ---------------------------------------------------------------------------
// Shared epilogue: out = acc + bias[n] + dot8(t[m], loraB[n])
// ---------------------------------------------------------------------------
__device__ __forceinline__ void gemm_epilogue(
    v8f (&acc)[4][2], int m0, int n0, int wm, int wn, int lg, int lh,
    const float* __restrict__ bias, const float* __restrict__ lB,
    const float* __restrict__ t, float* __restrict__ out) {
#pragma unroll
    for (int mt = 0; mt < 4; ++mt) {
#pragma unroll
        for (int nt = 0; nt < 2; ++nt) {
            const int col = n0 + wn + nt * 16 + lg;
            const float bb = bias[col];
            const float4 lb0 = *(const float4*)(lB + (size_t)col * R_LORA);
            const float4 lb1 = *(const float4*)(lB + (size_t)col * R_LORA + 4);
#pragma unroll
            for (int v = 0; v < 8; ++v) {
                const int rowg = m0 + wm + mt * 16 + v + lh * 8;
                const float* tr = t + (size_t)rowg * R_LORA;
                const float4 t0 = *(const float4*)tr;
                const float4 t1 = *(const float4*)(tr + 4);
                const float lora = t0.x * lb0.x + t0.y * lb0.y +
                                   t0.z * lb0.z + t0.w * lb0.w +
                                   t1.x * lb1.x + t1.y * lb1.y +
                                   t1.z * lb1.z + t1.w * lb1.w;
                out[(size_t)rowg * N_TOT + col] = acc[mt][nt][v] + bb + lora;
            }
        }
    }
}

// ---------------------------------------------------------------------------
// Primary GEMM: pre-split bf16 operands, async global->LDS, double buffer.
// 8 waves (2x4); each wave owns 64x32 of C (4x2 wmma tiles).
// ---------------------------------------------------------------------------
__global__ __launch_bounds__(256) void lora_gemm_async(
    const u16* __restrict__ xh, const u16* __restrict__ xl,
    const u16* __restrict__ wh, const u16* __restrict__ wl,
    const float* __restrict__ bias, const float* __restrict__ lB,
    const float* __restrict__ t, float* __restrict__ out) {

    __shared__ u16 sXh[2][TILE_E];
    __shared__ u16 sXl[2][TILE_E];
    __shared__ u16 sWh[2][TILE_E];
    __shared__ u16 sWl[2][TILE_E];

    const int tid  = threadIdx.x;
    const int lane = tid & 31;
    const int wave = tid >> 5;
    const int wm   = (wave & 1) * 64;
    const int wn   = (wave >> 1) * 32;
    const int lg   = lane & 15;
    const int lh   = lane >> 4;

    const int m0 = blockIdx.y * BM;
    const int n0 = blockIdx.x * BN;

    // Each thread owns 2 b128 (8 u16) slots per operand array.
    const int rbase8 = tid >> 2;   // 0..63 (+64 for i=1)
    const int c8     = tid & 3;

    const u16* pxh[2]; const u16* pxl[2]; const u16* pwh[2]; const u16* pwl[2];
    unsigned oxh[2], oxl[2], owh[2], owl[2];   // LDS byte offsets (buffer 0)
#pragma unroll
    for (int i = 0; i < 2; ++i) {
        const int row = rbase8 + i * 64;
        const size_t gx = (size_t)(m0 + row) * K_TOT + c8 * 8;
        const size_t gw = (size_t)(n0 + row) * K_TOT + c8 * 8;
        pxh[i] = xh + gx; pxl[i] = xl + gx;
        pwh[i] = wh + gw; pwl[i] = wl + gw;
        const unsigned idx = (unsigned)(row * LDK + c8 * 8);
        oxh[i] = (unsigned)(size_t)&sXh[0][idx];
        oxl[i] = (unsigned)(size_t)&sXl[0][idx];
        owh[i] = (unsigned)(size_t)&sWh[0][idx];
        owl[i] = (unsigned)(size_t)&sWl[0][idx];
    }

    auto prefetch = [&](int buf) {
        const unsigned boff = (unsigned)buf * TILE_B;
#pragma unroll
        for (int i = 0; i < 2; ++i) {
            async_copy_b128(oxh[i] + boff, pxh[i]);
            async_copy_b128(oxl[i] + boff, pxl[i]);
            async_copy_b128(owh[i] + boff, pwh[i]);
            async_copy_b128(owl[i] + boff, pwl[i]);
            pxh[i] += BK; pxl[i] += BK; pwh[i] += BK; pwl[i] += BK;
        }
    };

    v8f acc[4][2] = {};

    prefetch(0);
    wait_async0();
    __syncthreads();

    for (int kt = 0; kt < KT; ++kt) {
        const int buf = kt & 1;
        const bool hasNext = (kt + 1 < KT);
        if (hasNext) prefetch(buf ^ 1);

        const u16* bXh = &sXh[buf][0];
        const u16* bXl = &sXl[buf][0];
        const u16* bWh = &sWh[buf][0];
        const u16* bWl = &sWl[buf][0];

        v16bf bh[2], bl[2];
#pragma unroll
        for (int nt = 0; nt < 2; ++nt) {
            bh[nt] = loadB_frag(bWh, wn, nt, lg, lh);
            bl[nt] = loadB_frag(bWl, wn, nt, lg, lh);
        }
#pragma unroll
        for (int mt = 0; mt < 4; ++mt) {
            v16bf ah = loadA_frag(bXh, wm, mt, lg, lh);
            v16bf al = loadA_frag(bXl, wm, mt, lg, lh);
#pragma unroll
            for (int nt = 0; nt < 2; ++nt) {
                acc[mt][nt] = __builtin_amdgcn_wmma_f32_16x16x32_bf16(
                    false, ah, false, bh[nt], (short)0, acc[mt][nt], false, false);
                acc[mt][nt] = __builtin_amdgcn_wmma_f32_16x16x32_bf16(
                    false, ah, false, bl[nt], (short)0, acc[mt][nt], false, false);
                acc[mt][nt] = __builtin_amdgcn_wmma_f32_16x16x32_bf16(
                    false, al, false, bh[nt], (short)0, acc[mt][nt], false, false);
            }
        }

        if (hasNext) {
            wait_async0();
            __syncthreads();
        }
    }

    gemm_epilogue(acc, m0, n0, wm, wn, lg, lh, bias, lB, t, out);
}

// ---------------------------------------------------------------------------
// Fallback GEMM (ws too small for full pre-split): register-staged tiles,
// on-the-fly fp32 -> bf16 hi/lo split for non-pre operands.
// ---------------------------------------------------------------------------
template <bool WPRE>
__global__ __launch_bounds__(256) void lora_gemm_fallback(
    const float* __restrict__ xf,
    const float* __restrict__ wf, const u16* __restrict__ wh, const u16* __restrict__ wl,
    const float* __restrict__ bias, const float* __restrict__ lB,
    const float* __restrict__ t, float* __restrict__ out) {

    __shared__ u16 sXh[TILE_E];
    __shared__ u16 sXl[TILE_E];
    __shared__ u16 sWh[TILE_E];
    __shared__ u16 sWl[TILE_E];

    const int tid  = threadIdx.x;
    const int lane = tid & 31;
    const int wave = tid >> 5;
    const int wm   = (wave & 1) * 64;
    const int wn   = (wave >> 1) * 32;
    const int lg   = lane & 15;
    const int lh   = lane >> 4;

    const int m0 = blockIdx.y * BM;
    const int n0 = blockIdx.x * BN;

    const int rbase4 = tid >> 3, c4 = tid & 7;   // fp32: 4 x float4 / thread
    const int rbase8 = tid >> 2, c8 = tid & 3;   // bf16: 2 x uint4 / thread

    float4 rx[4], rw[4];
    uint4  rwh[2], rwl[2];

    auto loadTiles = [&](int k0) {
#pragma unroll
        for (int i = 0; i < 4; ++i)
            rx[i] = *(const float4*)(xf + (size_t)(m0 + rbase4 + i * 32) * K_TOT + k0 + c4 * 4);
        if constexpr (WPRE) {
#pragma unroll
            for (int i = 0; i < 2; ++i) {
                size_t off = (size_t)(n0 + rbase8 + i * 64) * K_TOT + k0 + c8 * 8;
                rwh[i] = *(const uint4*)(wh + off);
                rwl[i] = *(const uint4*)(wl + off);
            }
        } else {
#pragma unroll
            for (int i = 0; i < 4; ++i)
                rw[i] = *(const float4*)(wf + (size_t)(n0 + rbase4 + i * 32) * K_TOT + k0 + c4 * 4);
        }
    };

    auto stageTiles = [&]() {
#pragma unroll
        for (int i = 0; i < 4; ++i) {
            int idx = (rbase4 + i * 32) * LDK + c4 * 4;
            uint2 hv, lv;
            split4(rx[i], hv, lv);
            *(uint2*)&sXh[idx] = hv;
            *(uint2*)&sXl[idx] = lv;
        }
        if constexpr (WPRE) {
#pragma unroll
            for (int i = 0; i < 2; ++i) {
                int idx = (rbase8 + i * 64) * LDK + c8 * 8;
                *(uint4*)&sWh[idx] = rwh[i];
                *(uint4*)&sWl[idx] = rwl[i];
            }
        } else {
#pragma unroll
            for (int i = 0; i < 4; ++i) {
                int idx = (rbase4 + i * 32) * LDK + c4 * 4;
                uint2 hv, lv;
                split4(rw[i], hv, lv);
                *(uint2*)&sWh[idx] = hv;
                *(uint2*)&sWl[idx] = lv;
            }
        }
    };

    v8f acc[4][2] = {};

    loadTiles(0);
    stageTiles();
    __syncthreads();

    for (int kt = 0; kt < KT; ++kt) {
        const bool hasNext = (kt + 1 < KT);
        if (hasNext) loadTiles((kt + 1) * BK);

        v16bf bh[2], bl[2];
#pragma unroll
        for (int nt = 0; nt < 2; ++nt) {
            bh[nt] = loadB_frag(sWh, wn, nt, lg, lh);
            bl[nt] = loadB_frag(sWl, wn, nt, lg, lh);
        }
#pragma unroll
        for (int mt = 0; mt < 4; ++mt) {
            v16bf ah = loadA_frag(sXh, wm, mt, lg, lh);
            v16bf al = loadA_frag(sXl, wm, mt, lg, lh);
#pragma unroll
            for (int nt = 0; nt < 2; ++nt) {
                acc[mt][nt] = __builtin_amdgcn_wmma_f32_16x16x32_bf16(
                    false, ah, false, bh[nt], (short)0, acc[mt][nt], false, false);
                acc[mt][nt] = __builtin_amdgcn_wmma_f32_16x16x32_bf16(
                    false, ah, false, bl[nt], (short)0, acc[mt][nt], false, false);
                acc[mt][nt] = __builtin_amdgcn_wmma_f32_16x16x32_bf16(
                    false, al, false, bh[nt], (short)0, acc[mt][nt], false, false);
            }
        }

        if (hasNext) {
            __syncthreads();
            stageTiles();
            __syncthreads();
        }
    }

    gemm_epilogue(acc, m0, n0, wm, wn, lg, lh, bias, lB, t, out);
}

extern "C" void kernel_launch(void* const* d_in, const int* in_sizes, int n_in,
                              void* d_out, int out_size, void* d_ws, size_t ws_size,
                              hipStream_t stream) {
    (void)in_sizes; (void)n_in; (void)out_size;
    const float* x   = (const float*)d_in[0];
    const int*   ids = (const int*)d_in[1];
    const float* W   = (const float*)d_in[2];
    const float* b   = (const float*)d_in[3];
    const float* lA  = (const float*)d_in[4];
    const float* lB  = (const float*)d_in[5];
    float* out = (float*)d_out;

    const size_t T_BYTES = (size_t)M_TOT * R_LORA * sizeof(float);  // 512 KB
    const size_t W_HALF  = (size_t)N_TOT * K_TOT * sizeof(u16);     // 32 MB
    const size_t X_HALF  = (size_t)M_TOT * K_TOT * sizeof(u16);     // 128 MB

    char* p = (char*)d_ws;
    float* t  = (float*)p;  p += T_BYTES;
    u16* w_hi = (u16*)p;    p += W_HALF;
    u16* w_lo = (u16*)p;    p += W_HALF;
    u16* x_hi = (u16*)p;    p += X_HALF;
    u16* x_lo = (u16*)p;

    const bool wpre = ws_size >= T_BYTES + 2 * W_HALF;
    const bool xpre = ws_size >= T_BYTES + 2 * W_HALF + 2 * X_HALF;

    lora_t_kernel<<<dim3(M_TOT), dim3(256), 0, stream>>>(x, ids, lA, t);

    if (wpre) {
        const long long nW8 = (long long)N_TOT * K_TOT / 8;
        split_bf16_kernel<<<dim3((unsigned)(nW8 / 256)), dim3(256), 0, stream>>>(
            W, w_hi, w_lo, nW8);
    }
    if (xpre) {
        const long long nX8 = (long long)M_TOT * K_TOT / 8;
        split_bf16_kernel<<<dim3((unsigned)(nX8 / 256)), dim3(256), 0, stream>>>(
            x, x_hi, x_lo, nX8);
    }

    const dim3 grid(N_TOT / BN, M_TOT / BM);
    if (xpre && wpre) {
        lora_gemm_async<<<grid, dim3(256), 0, stream>>>(
            x_hi, x_lo, w_hi, w_lo, b, lB, t, out);
    } else if (wpre) {
        lora_gemm_fallback<true><<<grid, dim3(256), 0, stream>>>(
            x, W, w_hi, w_lo, b, lB, t, out);
    } else {
        lora_gemm_fallback<false><<<grid, dim3(256), 0, stream>>>(
            x, W, w_hi, w_lo, b, lB, t, out);
    }
}
